// AttentionPreparer_64158221467716
// MI455X (gfx1250) — matrix operand order repaired and checked by reading, not compiled, not run on hardware
//
#include <hip/hip_runtime.h>
#include <math.h>

// ---------------- problem constants ----------------
#define T_DIM   4096
#define HID     7168
#define NHEAD   128
#define D_NOPE  128
#define D_ROPE  64
#define Q_LORA  1536
#define KV_LORA 512
#define D_QK    192              // D_NOPE + D_ROPE
#define FUSED_N 2112             // Q_LORA + KV_LORA + D_ROPE
#define QB_N    24576            // NHEAD * D_QK
#define LOG2_10000 13.287712379549449f

typedef __attribute__((ext_vector_type(16))) __bf16 v16bf;
typedef __attribute__((ext_vector_type(8)))  __bf16 v8bf;
typedef __attribute__((ext_vector_type(8)))  float  v8f;

// ---------------- tiled bf16 WMMA GEMM ----------------
// Block tile 128x128, K-step 32, 256 threads = 8 waves (4 along M x 2 along N),
// each wave computes 32x64 via 2x4 v_wmma_f32_16x16x32_bf16 accumulators.
// A: f32 (converted at LDS fill) or bf16.  B: always f32 [K,N] row-major,
// converted + transposed into LDS (K contiguous per column).
// EPI 0: plain f32 store (ldc).
// EPI 1: GEMM2 split: per-head nope span -> bf16 ws; pe span -> RoPE -> f32 out.
// BOUNDS: N bounds handling (clamped loads, guarded stores) -- GEMM1 only.
#define LDS_STRIDE 48            // bf16 elems per LDS row (96B, 16B aligned)

template <int EPI, bool AF32, bool BOUNDS>
__global__ __launch_bounds__(256) void gemm_wmma(
    const void* __restrict__ Aptr, const float* __restrict__ B,
    float* __restrict__ Cf, __bf16* __restrict__ Cb,
    const int* __restrict__ positions,
    int N, int K, int lda, int ldb, int ldc,
    long aBatch, long bBatch, long cBatch)
{
    __shared__ __attribute__((aligned(16))) __bf16 As[128 * LDS_STRIDE];
    __shared__ __attribute__((aligned(16))) __bf16 Bs[128 * LDS_STRIDE];

    const int tid  = threadIdx.x;
    const int lane = tid & 31;
    const int wv   = tid >> 5;
    const int wm   = wv & 3;          // 0..3 : 32-row slice
    const int wn   = wv >> 2;         // 0..1 : 64-col slice
    const int lo   = lane & 15;
    const int hi   = lane >> 4;

    const int n0 = blockIdx.x * 128;
    const int m0 = blockIdx.y * 128;
    const long aOff = (long)blockIdx.z * aBatch;
    const long bOff = (long)blockIdx.z * bBatch;
    const long cOff = (long)blockIdx.z * cBatch;

    v8f acc[2][4];
#pragma unroll
    for (int mi = 0; mi < 2; ++mi)
#pragma unroll
        for (int ni = 0; ni < 4; ++ni) acc[mi][ni] = (v8f)0.f;

    // loader thread mapping (constant over K loop)
    const int ar  = tid >> 1;             // A: row 0..127
    const int ahk = (tid & 1) * 16;       // A: k sub-block 0/16
    const int bnl = tid & 127;            // B: local col 0..127
    const int bkg = tid >> 7;             // B: k group 0/1 (16 k each)
    int ncol = n0 + bnl;
    if (BOUNDS) ncol = (ncol < N) ? ncol : (N - 1);   // clamp: OOB cols never stored

    // hoisted base pointers (advance per K-step; no in-loop 64-bit muls)
    const float*  Af = AF32 ? (const float*)Aptr + aOff + (long)(m0 + ar) * lda + ahk : nullptr;
    const __bf16* Ab = AF32 ? nullptr : (const __bf16*)Aptr + aOff + (long)(m0 + ar) * lda + ahk;
    const float*  Bp = B + bOff + (long)(bkg * 16) * ldb + ncol;
    const long bStep = (long)32 * ldb;
    __bf16* asDst = As + ar * LDS_STRIDE + ahk;
    __bf16* bsDst = Bs + bnl * LDS_STRIDE + bkg * 16;
    const __bf16* aSrc = As + (wm * 32 + lo) * LDS_STRIDE + hi * 16;
    const __bf16* bSrc = Bs + (wn * 64 + lo) * LDS_STRIDE + hi * 16;

    for (int k0 = 0; k0 < K; k0 += 32) {
        // ---- A tile -> LDS (bf16) ----
        if (AF32) {
            const float4* g = (const float4*)Af;
            v8bf t0, t1;
#pragma unroll
            for (int v = 0; v < 2; ++v) {
                float4 f = g[v];
                t0[v * 4 + 0] = (__bf16)f.x; t0[v * 4 + 1] = (__bf16)f.y;
                t0[v * 4 + 2] = (__bf16)f.z; t0[v * 4 + 3] = (__bf16)f.w;
            }
#pragma unroll
            for (int v = 0; v < 2; ++v) {
                float4 f = g[2 + v];
                t1[v * 4 + 0] = (__bf16)f.x; t1[v * 4 + 1] = (__bf16)f.y;
                t1[v * 4 + 2] = (__bf16)f.z; t1[v * 4 + 3] = (__bf16)f.w;
            }
            *(v8bf*)asDst       = t0;
            *(v8bf*)(asDst + 8) = t1;
            Af += 32;
        } else {
            *(v8bf*)asDst       = *(const v8bf*)Ab;
            *(v8bf*)(asDst + 8) = *(const v8bf*)(Ab + 8);
            Ab += 32;
        }
        // ---- B tile -> LDS transposed (K contiguous per column) ----
        {
            v8bf t0, t1;
#pragma unroll
            for (int j = 0; j < 8; ++j) t0[j] = (__bf16)Bp[(long)j * ldb];
#pragma unroll
            for (int j = 0; j < 8; ++j) t1[j] = (__bf16)Bp[(long)(j + 8) * ldb];
            *(v8bf*)bsDst       = t0;
            *(v8bf*)(bsDst + 8) = t1;
            Bp += bStep;
        }
        __syncthreads();

        // ---- fragments + WMMA ----
        v16bf afr[2];
#pragma unroll
        for (int mi = 0; mi < 2; ++mi) {
            const __bf16* s = aSrc + mi * 16 * LDS_STRIDE;
            // lane half hi: K = hi*16 + [0..15]
            *(v8bf*)&afr[mi]       = *(const v8bf*)s;
            *((v8bf*)&afr[mi] + 1) = *(const v8bf*)(s + 8);
        }
        v16bf bfr[4];
#pragma unroll
        for (int ni = 0; ni < 4; ++ni) {
            const __bf16* s = bSrc + ni * 16 * LDS_STRIDE;
            // lane half hi: K = hi*16 + [0..15], contiguous in transposed LDS
            *(v8bf*)&bfr[ni]       = *(const v8bf*)s;
            *((v8bf*)&bfr[ni] + 1) = *(const v8bf*)(s + 8);
        }
#pragma unroll
        for (int mi = 0; mi < 2; ++mi)
#pragma unroll
            for (int ni = 0; ni < 4; ++ni)
                acc[mi][ni] = __builtin_amdgcn_wmma_f32_16x16x32_bf16(
                    false, afr[mi], false, bfr[ni], (short)0, acc[mi][ni], false, false);
        __syncthreads();
    }

    // ---- epilogue ----  C/D layout: row = hi*8 + d, col = lo (per 16x16 tile)
    if (EPI == 0) {
#pragma unroll
        for (int mi = 0; mi < 2; ++mi)
#pragma unroll
            for (int ni = 0; ni < 4; ++ni) {
                const int col = n0 + wn * 64 + ni * 16 + lo;
#pragma unroll
                for (int d = 0; d < 8; ++d) {
                    const int row = m0 + wm * 32 + mi * 16 + hi * 8 + d;
                    if (!BOUNDS || col < N)
                        Cf[cOff + (long)row * ldc + col] = acc[mi][ni][d];
                }
            }
    } else {
        // GEMM2 split. Each wave's 64-col span is 64-aligned; every head's pe
        // region [h*192+128, h*192+192) is exactly one such span, so the span
        // is uniformly all-nope or all-pe.
        const int span0 = n0 + wn * 64;
        const int head  = span0 / D_QK;
        const int soff  = span0 - head * D_QK;     // 0, 64 (nope) or 128 (pe)
        if (soff != D_NOPE) {
            // nope: store bf16 into ws laid out [T, H*128] for GEMM3
#pragma unroll
            for (int mi = 0; mi < 2; ++mi)
#pragma unroll
                for (int ni = 0; ni < 4; ++ni) {
                    const int cih = soff + ni * 16 + lo;
#pragma unroll
                    for (int d = 0; d < 8; ++d) {
                        const int row = m0 + wm * 32 + mi * 16 + hi * 8 + d;
                        Cb[(long)row * (NHEAD * D_NOPE) + head * D_NOPE + cih] =
                            (__bf16)acc[mi][ni][d];
                    }
                }
        } else {
            // pe: rope pair (j, j+32): x1 in acc[.][ni], x2 in acc[.][ni+2], same lane
#pragma unroll
            for (int ni = 0; ni < 2; ++ni) {
                const int j = ni * 16 + lo;
                const float inv = exp2f(-(float)j * (LOG2_10000 / 32.f));
#pragma unroll
                for (int mi = 0; mi < 2; ++mi)
#pragma unroll
                    for (int d = 0; d < 8; ++d) {
                        const int row = m0 + wm * 32 + mi * 16 + hi * 8 + d;
                        float s, c;
                        sincosf((float)positions[row] * inv, &s, &c);
                        const float x1 = acc[mi][ni][d];
                        const float x2 = acc[mi][ni + 2][d];
                        float* p = Cf + (long)row * (NHEAD * D_ROPE) + head * D_ROPE;
                        p[j]      = x1 * c - x2 * s;
                        p[j + 32] = x2 * c + x1 * s;
                    }
            }
        }
    }
}

// ---------------- rmsnorm + k outputs ----------------
__device__ inline float block_sum256(float v, float* sh) {
#pragma unroll
    for (int off = 16; off > 0; off >>= 1) v += __shfl_down(v, off, 32);
    const int lane = threadIdx.x & 31, w = threadIdx.x >> 5;
    __syncthreads();
    if (lane == 0) sh[w] = v;
    __syncthreads();
    float tot = 0.f;
#pragma unroll
    for (int i = 0; i < 8; ++i) tot += sh[i];
    return tot;
}

__global__ __launch_bounds__(256) void rmsnorm_kernel(
    const float* __restrict__ fused, const float* __restrict__ q_ln_w,
    const float* __restrict__ kv_ln_w, const int* __restrict__ positions,
    __bf16* __restrict__ qlat_norm, float* __restrict__ out_knope,
    float* __restrict__ out_kpe)
{
    __shared__ float sh[8];
    const int t = blockIdx.x;
    const float* row = fused + (long)t * FUSED_N;

    float s1 = 0.f, s2 = 0.f;
    for (int i = threadIdx.x; i < Q_LORA; i += 256) { float v = row[i]; s1 += v * v; }
    for (int i = threadIdx.x; i < KV_LORA; i += 256) { float v = row[Q_LORA + i]; s2 += v * v; }
    const float r1 = rsqrtf(block_sum256(s1, sh) * (1.f / Q_LORA) + 1e-6f);
    const float r2 = rsqrtf(block_sum256(s2, sh) * (1.f / KV_LORA) + 1e-6f);

    for (int i = threadIdx.x; i < Q_LORA; i += 256)
        qlat_norm[(long)t * Q_LORA + i] = (__bf16)(row[i] * r1 * q_ln_w[i]);
    for (int i = threadIdx.x; i < KV_LORA; i += 256)
        out_knope[(long)t * KV_LORA + i] = row[Q_LORA + i] * r2 * kv_ln_w[i];

    // k_pe RoPE: 32 pairs
    if (threadIdx.x < 32) {
        const int j = threadIdx.x;
        const float x1 = row[Q_LORA + KV_LORA + j];
        const float x2 = row[Q_LORA + KV_LORA + 32 + j];
        const float inv = exp2f(-(float)j * (LOG2_10000 / 32.f));
        float s, c;
        sincosf((float)positions[t] * inv, &s, &c);
        out_kpe[(long)t * D_ROPE + j]      = x1 * c - x2 * s;
        out_kpe[(long)t * D_ROPE + 32 + j] = x2 * c + x1 * s;
    }
}

// ---------------- launch ----------------
extern "C" void kernel_launch(void* const* d_in, const int* in_sizes, int n_in,
                              void* d_out, int out_size, void* d_ws, size_t ws_size,
                              hipStream_t stream)
{
    const int*   positions = (const int*)d_in[0];
    const float* hidden    = (const float*)d_in[1];
    const float* w_fused   = (const float*)d_in[2];
    const float* q_ln_w    = (const float*)d_in[3];
    const float* w_qb      = (const float*)d_in[4];
    const float* kv_ln_w   = (const float*)d_in[5];
    const float* w_kc      = (const float*)d_in[6];
    float* out = (float*)d_out;

    // output regions (floats), in return order (q_pe, k_pe, q_nope_out, k_nope)
    float* out_qpe   = out;                                        // T*H*64
    float* out_kpe   = out + (long)T_DIM * NHEAD * D_ROPE;         // T*64
    float* out_qnope = out_kpe + (long)T_DIM * D_ROPE;             // T*H*512
    float* out_knope = out_qnope + (long)T_DIM * NHEAD * KV_LORA;  // T*512

    // workspace: fused f32 (34.6MB) | qlat bf16 (12.6MB) | qnope bf16 (134MB)
    char* ws = (char*)d_ws;
    float*  fused = (float*)ws;
    __bf16* qlat  = (__bf16*)(ws + (size_t)T_DIM * FUSED_N * 4);
    __bf16* qnope = (__bf16*)((char*)qlat + (size_t)T_DIM * Q_LORA * 2);

    const dim3 blk(256);

    // GEMM1: hidden[4096,7168] x w_fused[7168,2112] -> fused (f32, ws)
    gemm_wmma<0, true, true><<<dim3((FUSED_N + 127) / 128, T_DIM / 128, 1), blk, 0, stream>>>(
        hidden, w_fused, fused, nullptr, nullptr,
        FUSED_N, HID, HID, FUSED_N, FUSED_N, 0, 0, 0);

    // rmsnorms + k_nope + k_pe RoPE
    rmsnorm_kernel<<<T_DIM, blk, 0, stream>>>(fused, q_ln_w, kv_ln_w, positions,
                                              qlat, out_knope, out_kpe);

    // GEMM2: qlat_bf16[4096,1536] x w_qb[1536,24576]
    //  -> q_nope bf16 ws ([T, H*128]) + q_pe f32 out (roped in epilogue)
    gemm_wmma<1, false, false><<<dim3(QB_N / 128, T_DIM / 128, 1), blk, 0, stream>>>(
        qlat, w_qb, out_qpe, qnope, positions,
        QB_N, Q_LORA, Q_LORA, QB_N, 0, 0, 0, 0);

    // GEMM3 (batched over 128 heads): qnope[:,h*128:+128] x w_kc[h][128,512]
    gemm_wmma<0, false, false><<<dim3(KV_LORA / 128, T_DIM / 128, NHEAD), blk, 0, stream>>>(
        qnope, w_kc, out_qnope, nullptr, nullptr,
        KV_LORA, D_NOPE, NHEAD * D_NOPE, KV_LORA, NHEAD * KV_LORA,
        (long)D_NOPE, (long)D_NOPE * KV_LORA, (long)KV_LORA);
}
